// TAIGCN_14362370638523
// MI455X (gfx1250) — compile-verified, matrix operand order checked by reading, not run on hardware
//
#include <hip/hip_runtime.h>
#include <hip/hip_bf16.h>

// ---------------------------------------------------------------------------
// CDNA5 / gfx1250 TAGCN pipeline:
//   x = leaky(X @ W0 + b0); x = S@x; x = x @ W1 + b1; x = (S@)^3 x
//   s_emb = Sess @ x
// GEMMs: v_wmma_f32_16x16x32_bf16 (bf16 A/B, fp32 accum), 128x128 block tile,
// double-buffered LDS, software-pipelined unconditional (clamped) global
// loads so the 116 MB item_features stream stays at full HBM bandwidth.
// SpMMs: wave-per-row over sorted COO (deterministic), pipelined gathers.
// ---------------------------------------------------------------------------

typedef __attribute__((ext_vector_type(16))) __bf16 v16bf;
typedef __attribute__((ext_vector_type(8)))  __bf16 v8bf;
typedef __attribute__((ext_vector_type(4)))  __bf16 v4bf;
typedef __attribute__((ext_vector_type(8)))  float  v8f;

#define LDS_STRIDE 40   // bf16 elems per LDS row: 32 data + 8 pad.
                        // 80B stride -> 16 rows hit distinct 4-bank groups,
                        // fragment chunks stay 16B aligned.

// ---------------------------------------------------------------------------
// Fused GEMM: Y[M,N] = act(A[M,K] @ W[K,N] + bias)
// Block: 256 threads = 8 waves; block tile 128(M) x 128(N); BK = 32.
// Wave grid 4(M) x 2(N); wave tile 32x64 = 2x4 wmma 16x16 tiles.
// Double-buffered LDS, one barrier per K-step, register-staged prefetch.
// ---------------------------------------------------------------------------
template<bool LEAKY>
__global__ __launch_bounds__(256)
void gemm_bias_act_wmma(const float* __restrict__ A,
                        const float* __restrict__ W,
                        const float* __restrict__ bias,
                        float* __restrict__ Y,
                        int M, int N, int K)
{
    __shared__ __bf16 lA[2][128 * LDS_STRIDE];   // A tile, row-major [m][k]
    __shared__ __bf16 lB[2][128 * LDS_STRIDE];   // W tile, transposed [n][k]

    const int t     = threadIdx.x;
    const int lane  = t & 31;
    const int wave  = t >> 5;
    const int wm    = wave & 3;             // wave M index (0..3) -> 32*wm
    const int wn    = wave >> 2;            // wave N index (0..1) -> 64*wn
    const int l     = lane & 15;
    const int khalf = (lane >> 4) << 3;     // 0/8: fragment K-phase & C M-phase

    const int mBlock = blockIdx.x * 128;
    const int nBlock = blockIdx.y * 128;

    // --- per-thread tile-load geometry (all unconditional, clamped) ---
    // A tile: 128 rows x 32 k = 1024 float4 slots, 4 per thread.
    const int aRow[1]  = {};  (void)aRow;
    // W tile: 32 k x 128 n = 1024 float4 slots, 4 per thread.

    v8f acc[2][4] = {};
    float4 ra[4], rw[4];

    const int ksteps = (K + 31) >> 5;

    // ---- load tile kt into registers (clamped addresses, cndmask zeroing) --
    auto load_tile = [&](int kt) {
        const int k0 = kt << 5;
        #pragma unroll
        for (int i = 0; i < 4; ++i) {
            int idx  = t + (i << 8);            // 0..1023
            int row  = idx >> 3;                // 0..127
            int kc   = (idx & 7) << 2;          // 0..28
            int gr   = mBlock + row;
            bool okm = gr < M;
            bool okk = (k0 + kc) < K;
            int grc  = okm ? gr : (M - 1);
            int kac  = okk ? (k0 + kc) : 0;
            float4 v = *(const float4*)(A + (size_t)grc * K + kac);
            bool ok  = okm && okk;
            ra[i].x = ok ? v.x : 0.f;  ra[i].y = ok ? v.y : 0.f;
            ra[i].z = ok ? v.z : 0.f;  ra[i].w = ok ? v.w : 0.f;
        }
        #pragma unroll
        for (int i = 0; i < 4; ++i) {
            int idx  = t + (i << 8);            // 0..1023
            int kr   = idx >> 5;                // 0..31
            int nc   = (idx & 31) << 2;         // 0..124
            bool okk = (k0 + kr) < K;
            int kac  = okk ? (k0 + kr) : 0;
            float4 v = *(const float4*)(W + (size_t)kac * N + nBlock + nc);
            rw[i].x = okk ? v.x : 0.f;  rw[i].y = okk ? v.y : 0.f;
            rw[i].z = okk ? v.z : 0.f;  rw[i].w = okk ? v.w : 0.f;
        }
    };

    // ---- convert + store registers into LDS buffer p ----
    auto store_tile = [&](int p) {
        #pragma unroll
        for (int i = 0; i < 4; ++i) {
            int idx = t + (i << 8);
            int row = idx >> 3;
            int kc  = (idx & 7) << 2;
            v4bf b  = { (__bf16)ra[i].x, (__bf16)ra[i].y,
                        (__bf16)ra[i].z, (__bf16)ra[i].w };
            *(v4bf*)&lA[p][row * LDS_STRIDE + kc] = b;   // 8B packed store
        }
        #pragma unroll
        for (int i = 0; i < 4; ++i) {
            int idx = t + (i << 8);
            int kr  = idx >> 5;
            int nc  = (idx & 31) << 2;
            lB[p][(nc + 0) * LDS_STRIDE + kr] = (__bf16)rw[i].x;
            lB[p][(nc + 1) * LDS_STRIDE + kr] = (__bf16)rw[i].y;
            lB[p][(nc + 2) * LDS_STRIDE + kr] = (__bf16)rw[i].z;
            lB[p][(nc + 3) * LDS_STRIDE + kr] = (__bf16)rw[i].w;
        }
    };

    load_tile(0);
    store_tile(0);
    __syncthreads();

    for (int kt = 0; kt < ksteps; ++kt) {
        const int p = kt & 1;
        // issue next tile's global loads before computing (overlap with WMMA)
        if (kt + 1 < ksteps) load_tile(kt + 1);

        // ---- fragments per 16-bit A-matrix layout (ISA 7.12.2):
        // lanes 0-15: K 0..7 | 16..23 ; lanes 16-31: K 8..15 | 24..31;
        // each half is one contiguous 16B LDS chunk.
        union Frag { v16bf v; v8bf h[2]; };
        Frag afrag[2], bfrag[4];
        #pragma unroll
        for (int mi = 0; mi < 2; ++mi) {
            int row = wm * 32 + mi * 16 + l;
            afrag[mi].h[0] = *(const v8bf*)&lA[p][row * LDS_STRIDE + khalf];
            afrag[mi].h[1] = *(const v8bf*)&lA[p][row * LDS_STRIDE + khalf + 16];
        }
        #pragma unroll
        for (int ni = 0; ni < 4; ++ni) {
            int col = wn * 64 + ni * 16 + l;     // B fragment: lane -> N column
            bfrag[ni].h[0] = *(const v8bf*)&lB[p][col * LDS_STRIDE + khalf];
            bfrag[ni].h[1] = *(const v8bf*)&lB[p][col * LDS_STRIDE + khalf + 16];
        }
        #pragma unroll
        for (int mi = 0; mi < 2; ++mi)
            #pragma unroll
            for (int ni = 0; ni < 4; ++ni)
                acc[mi][ni] = __builtin_amdgcn_wmma_f32_16x16x32_bf16(
                    false, afrag[mi].v, false, bfrag[ni].v,
                    (short)0, acc[mi][ni], false, false);

        if (kt + 1 < ksteps) {
            store_tile((kt + 1) & 1);   // other buffer: no race with readers of p
            __syncthreads();
        }
    }

    // ---- epilogue: bias + optional LeakyReLU.
    // C layout: VGPR r -> M=r (lanes 0-15) / M=r+8 (lanes 16-31), N = lane&15.
    const int mc = mBlock + wm * 32;
    const int nc = nBlock + wn * 64;
    #pragma unroll
    for (int mi = 0; mi < 2; ++mi) {
        #pragma unroll
        for (int ni = 0; ni < 4; ++ni) {
            int col  = nc + ni * 16 + l;
            float bv = bias[col];
            #pragma unroll
            for (int r = 0; r < 8; ++r) {
                int row = mc + mi * 16 + r + khalf;
                if (row < M) {
                    float v = acc[mi][ni][r] + bv;
                    if (LEAKY) v = (v > 0.f) ? v : 0.01f * v;
                    Y[(size_t)row * N + col] = v;
                }
            }
        }
    }
}

// ---------------------------------------------------------------------------
// SpMM over sorted COO: Y[r,:] = sum_{e: row[e]==r} val[e] * X[col[e],:]
// One wave32 per output row (deterministic order). FJ4 = FEAT/128.
// Software-pipelined: next col/val loaded and gather row prefetched while
// the current gather is consumed.
// ---------------------------------------------------------------------------
__device__ inline int lower_bound_i32(const int* a, int n, int key) {
    int lo = 0, hi = n;
    while (lo < hi) { int mid = (lo + hi) >> 1; if (a[mid] < key) lo = mid + 1; else hi = mid; }
    return lo;
}

template<int FJ4>
__global__ __launch_bounds__(256)
void spmm_wave_row(const int* __restrict__ rows, const int* __restrict__ cols,
                   const float* __restrict__ vals, int nnz,
                   const float* __restrict__ X, float* __restrict__ Y, int n_rows)
{
    const int FEAT = FJ4 * 128;
    int r = blockIdx.x * 8 + (threadIdx.x >> 5);
    if (r >= n_rows) return;
    int lane = threadIdx.x & 31;

    int s = lower_bound_i32(rows, nnz, r);
    int e = lower_bound_i32(rows, nnz, r + 1);

    float4 acc[FJ4];
    #pragma unroll
    for (int j = 0; j < FJ4; ++j) acc[j] = make_float4(0.f, 0.f, 0.f, 0.f);

    if (s < e) {
        int   c = cols[s];
        float v = vals[s];
        for (int i = s; i < e; ++i) {
            int   cn = 0;
            float vn = 0.f;
            if (i + 1 < e) {
                cn = cols[i + 1];
                vn = vals[i + 1];
                __builtin_prefetch(X + (size_t)cn * FEAT + lane * 4, 0, 0);
            }
            #pragma unroll
            for (int j = 0; j < FJ4; ++j) {
                float4 x = *(const float4*)(X + (size_t)c * FEAT + lane * 4 + j * 128);
                acc[j].x += v * x.x; acc[j].y += v * x.y;
                acc[j].z += v * x.z; acc[j].w += v * x.w;
            }
            c = cn; v = vn;
        }
    }
    #pragma unroll
    for (int j = 0; j < FJ4; ++j)
        *(float4*)(Y + (size_t)r * FEAT + lane * 4 + j * 128) = acc[j];
}

// ---------------------------------------------------------------------------
// Inputs (setup_inputs order):
//  0 item_features [M,NFEAT] f32   1 W0 [NFEAT,256]   2 b0 [256]
//  3 W1 [256,128]                  4 b1 [128]
//  5 S_val [NNZ]                   6 sess_val [NINTER]
//  7 S_row [NNZ] (sorted)          8 S_col [NNZ]
//  9 sess_row [NINTER] (sorted)   10 sess_col [NINTER]  11 n_sess (device)
// d_out = s_emb [n_sess,128] ++ final_embeddings [M,128]
// ---------------------------------------------------------------------------
extern "C" void kernel_launch(void* const* d_in, const int* in_sizes, int n_in,
                              void* d_out, int out_size, void* d_ws, size_t ws_size,
                              hipStream_t stream)
{
    const float* item     = (const float*)d_in[0];
    const float* W0       = (const float*)d_in[1];
    const float* b0       = (const float*)d_in[2];
    const float* W1       = (const float*)d_in[3];
    const float* b1       = (const float*)d_in[4];
    const float* S_val    = (const float*)d_in[5];
    const float* sess_val = (const float*)d_in[6];
    const int*   S_row    = (const int*)d_in[7];
    const int*   S_col    = (const int*)d_in[8];
    const int*   sess_row = (const int*)d_in[9];
    const int*   sess_col = (const int*)d_in[10];

    const int H0     = in_sizes[2];               // 256
    const int H1     = in_sizes[4];               // 128
    const int NFEAT  = in_sizes[1] / H0;          // 968
    const int M      = in_sizes[0] / NFEAT;       // 30000
    const int NNZ    = in_sizes[5];               // 600000
    const int NINTER = in_sizes[6];               // 204800
    const int NSESS  = out_size / H1 - M;         // 4096 (host-derivable)

    // workspace: two M x H0 fp32 ping-pong buffers (61.4 MB)
    float* bufA = (float*)d_ws;
    float* bufB = bufA + (size_t)M * H0;

    float* out_s = (float*)d_out;                 // [NSESS, H1]
    float* out_f = out_s + (size_t)NSESS * H1;    // [M, H1] final embeddings

    dim3 blk(256);
    const int gM   = (M + 127) / 128;
    const int gRow = (M + 7) / 8;

    // x0 = leaky(item @ W0 + b0)
    gemm_bias_act_wmma<true><<<dim3(gM, H0 / 128), blk, 0, stream>>>(
        item, W0, b0, bufA, M, H0, NFEAT);
    // x1 = S @ x0                              (feat 256)
    spmm_wave_row<2><<<gRow, blk, 0, stream>>>(S_row, S_col, S_val, NNZ, bufA, bufB, M);
    // x2 = x1 @ W1 + b1
    gemm_bias_act_wmma<false><<<dim3(gM, H1 / 128), blk, 0, stream>>>(
        bufB, W1, b1, bufA, M, H1, H0);
    // x3 = S @ x2 ; x4 = S @ x3 ; final = S @ x4   (feat 128)
    spmm_wave_row<1><<<gRow, blk, 0, stream>>>(S_row, S_col, S_val, NNZ, bufA, bufB, M);
    spmm_wave_row<1><<<gRow, blk, 0, stream>>>(S_row, S_col, S_val, NNZ, bufB, bufA, M);
    spmm_wave_row<1><<<gRow, blk, 0, stream>>>(S_row, S_col, S_val, NNZ, bufA, out_f, M);
    // s_emb = Sess @ final
    spmm_wave_row<1><<<(NSESS + 7) / 8, blk, 0, stream>>>(
        sess_row, sess_col, sess_val, NINTER, out_f, out_s, NSESS);
}
